// LLMPITT2D_16329465659706
// MI455X (gfx1250) — matrix operand order repaired and checked by reading, not compiled, not run on hardware
//
#include <hip/hip_runtime.h>
#include <hip/hip_bf16.h>

#define BB     4
#define NN     16384
#define DD     128
#define HH     8
#define DHH    64
#define INNERD 512

typedef __attribute__((ext_vector_type(16))) _Float16 v16h;
typedef __attribute__((ext_vector_type(8)))  float    v8f;

__device__ __forceinline__ int lane_id() { return (int)(threadIdx.x & 31u); }

__device__ __forceinline__ v8f wmma_f16(v16h a, v16h b, v8f c) {
  // v_wmma_f32_16x16x32_f16: D = A(16x32) * B(32x16) + C(16x16 f32)
  return __builtin_amdgcn_wmma_f32_16x16x32_f16(false, a, false, b, (short)0, c, false, false);
}

// ---- Fragment loaders (ISA 7.12.2 layouts, wave32) -------------------------
// A 16x32 (MxK): lane half 0 -> K {0..7,16..23}, half 1 -> K {8..15,24..31}.
// Contiguous 8-element runs -> vectorizes to 16B loads.
template <typename T>
__device__ __forceinline__ v16h a_frag(const T* __restrict__ src, int ld, int m0, int k0) {
  const int l = lane_id(); const int half = l >> 4; const int m = m0 + (l & 15);
  const T* p = src + (size_t)m * ld + k0 + half * 8;
  v16h a;
#pragma unroll
  for (int e = 0; e < 8; ++e) a[e] = (_Float16)p[e];
#pragma unroll
  for (int e = 0; e < 8; ++e) a[8 + e] = (_Float16)p[16 + e];
  return a;
}

// B 32x16 (KxN) with B[k][n] = W[n][k] (W row-major [n][k]); contiguous reads.
// Same K order as A: half 0 -> K {0..7,16..23}, half 1 -> K {8..15,24..31}.
template <typename T>
__device__ __forceinline__ v16h bT_frag(const T* __restrict__ W, int ld, int k0, int n0) {
  const int l = lane_id(); const int half = l >> 4; const int n = n0 + (l & 15);
  const T* p = W + (size_t)n * ld + k0 + half * 8;
  v16h b;
#pragma unroll
  for (int e = 0; e < 8; ++e) b[e] = (_Float16)p[e];
#pragma unroll
  for (int e = 0; e < 8; ++e) b[8 + e] = (_Float16)p[16 + e];
  return b;
}

// C/D 16x16 f32 -> f16 LDS, stored TRANSPOSED: element (m,n) goes to dst[n][m].
__device__ __forceinline__ void c_storeT(_Float16* __restrict__ dst, int ld, int m0, int n0,
                                         v8f c, float s) {
  const int l = lane_id(); const int half = l >> 4; const int n = n0 + (l & 15);
#pragma unroll
  for (int r = 0; r < 8; ++r)
    dst[(size_t)n * ld + (m0 + half * 8 + r)] = (_Float16)(c[r] * s);
}

// C/D 16x16 f32 -> f16 LDS, row-major
__device__ __forceinline__ void c_store(_Float16* __restrict__ dst, int ld, int m0, int n0,
                                        v8f c, float s) {
  const int l = lane_id(); const int half = l >> 4; const int n = n0 + (l & 15);
#pragma unroll
  for (int r = 0; r < 8; ++r)
    dst[(size_t)(m0 + half * 8 + r) * ld + n] = (_Float16)(c[r] * s);
}

// Coalesced f32 global -> f16 LDS tile stage (nfloat4 = #float4 elements)
__device__ __forceinline__ void stage_tile(const float* __restrict__ g, _Float16* __restrict__ s,
                                           int nfloat4) {
  const float4* g4 = (const float4*)g;
  for (int i = (int)threadIdx.x; i < nfloat4; i += (int)blockDim.x) {
    float4 f = g4[i];
    _Float16* d = s + (size_t)i * 4;
    d[0] = (_Float16)f.x; d[1] = (_Float16)f.y;
    d[2] = (_Float16)f.z; d[3] = (_Float16)f.w;
  }
}

// ---------------------------------------------------------------------------
__global__ void k_zero_f32(float* __restrict__ p, int n) {
  int i = blockIdx.x * blockDim.x + threadIdx.x;
  if (i < n) p[i] = 0.0f;
}

// Vectorized f32 -> f16 conversion with optional scale (n must be multiple of 4)
__global__ void k_cvt_f16(const float* __restrict__ src, _Float16* __restrict__ dst,
                          int n4, float scale) {
  int i = blockIdx.x * blockDim.x + threadIdx.x;
  if (i < n4) {
    float4 f = ((const float4*)src)[i];
    _Float16* d = dst + (size_t)i * 4;
    d[0] = (_Float16)(f.x * scale); d[1] = (_Float16)(f.y * scale);
    d[2] = (_Float16)(f.z * scale); d[3] = (_Float16)(f.w * scale);
  }
}

// Kernel 1: K/V projection + dotsT[e][d] += (K^T V)^T accumulation.
// One wave per head; keys/values chunk staged once per block.
__global__ __launch_bounds__(256) void k_kv_dots(const float* __restrict__ keys,
                                                 const float* __restrict__ values,
                                                 const float* __restrict__ Wk,
                                                 float* __restrict__ dotsT) {
  __shared__ _Float16 kin[32][DD];        // 8 KB  staged keys chunk (f16)
  __shared__ _Float16 vin[32][DD];        // 8 KB  staged values chunk (f16)
  __shared__ _Float16 Kt[HH][DHH][32];    // 32 KB projected K, d-major
  __shared__ _Float16 Vt[HH][DHH][32];    // 32 KB projected V, e-major
  const int b = blockIdx.y;
  const int h = (int)(threadIdx.x >> 5);
  const float* kb = keys   + (size_t)b * NN * DD;
  const float* vb = values + (size_t)b * NN * DD;
  const float* Wh = Wk + (size_t)h * DHH * DD;   // head slice [64][128]
  _Float16* Kh = &Kt[h][0][0];
  _Float16* Vh = &Vt[h][0][0];

  // Hoist Wk B-fragments: invariant across all chunks (128 VGPRs)
  v16h wk[4][4];
#pragma unroll
  for (int dt = 0; dt < 4; ++dt)
#pragma unroll
    for (int ksi = 0; ksi < 4; ++ksi)
      wk[dt][ksi] = bT_frag(Wh, DD, ksi * 32, dt * 16);

  v8f acc[4][4];
#pragma unroll
  for (int i = 0; i < 4; ++i)
#pragma unroll
    for (int j = 0; j < 4; ++j) { v8f z = {}; acc[i][j] = z; }

  const int stride = (int)gridDim.x * 32;          // 32 blocks * 32 rows
  for (int n0 = (int)blockIdx.x * 32; n0 < NN; n0 += stride) {
    __syncthreads();   // previous iteration's readers of kin/vin are done
    stage_tile(kb + (size_t)n0 * DD, &kin[0][0], 32 * DD / 4);
    stage_tile(vb + (size_t)n0 * DD, &vin[0][0], 32 * DD / 4);
    if (n0 + stride < NN) {   // prefetch next chunk (global_prefetch_b8)
      __builtin_prefetch(kb + (size_t)(n0 + stride) * DD + threadIdx.x * 16, 0, 1);
      __builtin_prefetch(vb + (size_t)(n0 + stride) * DD + threadIdx.x * 16, 0, 1);
    }
    __syncthreads();

    // Projection: [32,128] @ Wk_h^T -> Kt/Vt (transposed, d-major)
#pragma unroll
    for (int nt = 0; nt < 2; ++nt) {
      v16h ak[4], av[4];
#pragma unroll
      for (int ksi = 0; ksi < 4; ++ksi) {
        ak[ksi] = a_frag(&kin[0][0], DD, nt * 16, ksi * 32);
        av[ksi] = a_frag(&vin[0][0], DD, nt * 16, ksi * 32);
      }
#pragma unroll
      for (int dt = 0; dt < 4; ++dt) {
        v8f ck = {}; v8f cv = {};
#pragma unroll
        for (int ksi = 0; ksi < 4; ++ksi) {
          ck = wmma_f16(ak[ksi], wk[dt][ksi], ck);
          cv = wmma_f16(av[ksi], wk[dt][ksi], cv);
        }
        c_storeT(Kh, 32, nt * 16, dt * 16, ck, 1.0f);
        c_storeT(Vh, 32, nt * 16, dt * 16, cv, 1.0f);
      }
    }
    // dots: A = K^T (d-major tile, contiguous), B = V (e-major via bT, contiguous)
    v16h bV[4];
#pragma unroll
    for (int et = 0; et < 4; ++et) bV[et] = bT_frag(Vh, 32, 0, et * 16);
#pragma unroll
    for (int dt = 0; dt < 4; ++dt) {
      v16h aT = a_frag(Kh, 32, dt * 16, 0);
#pragma unroll
      for (int et = 0; et < 4; ++et)
        acc[dt][et] = wmma_f16(aT, bV[et], acc[dt][et]);
    }
  }

  // Reduce partials into global dotsT[b][h][e][d] with fp32 atomics
  const int l = lane_id(); const int half = l >> 4; const int col = l & 15;
  float* dh = dotsT + ((size_t)b * HH + h) * DHH * DHH;
#pragma unroll
  for (int dt = 0; dt < 4; ++dt)
#pragma unroll
    for (int et = 0; et < 4; ++et)
#pragma unroll
      for (int r = 0; r < 8; ++r) {
        const int d = dt * 16 + half * 8 + r;
        const int e = et * 16 + col;
        __hip_atomic_fetch_add(&dh[e * DHH + d], acc[dt][et][r],
                               __ATOMIC_RELAXED, __HIP_MEMORY_SCOPE_AGENT);
      }
}

// Kernel 2: q projection, y = q @ (dots/N), out = y @ Wo^T + bo. 16 tokens/block.
// Weights and dots are pre-converted f16 (single b128 per fragment, no cvt).
__global__ __launch_bounds__(256) void k_out(const float* __restrict__ queries,
                                             const _Float16* __restrict__ Wq16,
                                             const _Float16* __restrict__ dots16,
                                             const _Float16* __restrict__ Wo16,
                                             const float* __restrict__ bo,
                                             float* __restrict__ out) {
  __shared__ _Float16 qin[16][DD];        // 4 KB staged query tile
  __shared__ _Float16 qlds[16][INNERD];   // 16 KB
  __shared__ _Float16 ylds[16][INNERD];   // 16 KB
  const int b  = blockIdx.y;
  const int n0 = (int)blockIdx.x * 16;
  const int h  = (int)(threadIdx.x >> 5);
  const float*    qb  = queries + (size_t)b * NN * DD;
  const _Float16* Wqh = Wq16 + (size_t)h * DHH * DD;
  const _Float16* dh  = dots16 + ((size_t)b * HH + h) * DHH * DHH;   // [e][d], 1/N folded

  stage_tile(qb + (size_t)n0 * DD, &qin[0][0], 16 * DD / 4);
  __syncthreads();

  // q_h[16,64] = q_tile[16,128] @ Wq_h^T
#pragma unroll
  for (int dt = 0; dt < 4; ++dt) {
    v8f c = {};
#pragma unroll
    for (int ksi = 0; ksi < 4; ++ksi) {
      v16h a  = a_frag(&qin[0][0], DD, 0, ksi * 32);
      v16h bw = bT_frag(Wqh, DD, ksi * 32, dt * 16);
      c = wmma_f16(a, bw, c);
    }
    c_store(&qlds[0][0], INNERD, 0, h * DHH + dt * 16, c, 1.0f);
  }
  __syncthreads();

  // y_h[16,64] = q_h @ (dots_h/N)   (dots stored [e][d] f16 -> bT_frag contiguous)
#pragma unroll
  for (int et = 0; et < 4; ++et) {
    v8f c = {};
#pragma unroll
    for (int ksi = 0; ksi < 2; ++ksi) {
      v16h a  = a_frag(&qlds[0][0], INNERD, 0, h * DHH + ksi * 32);
      v16h bd = bT_frag(dh, DHH, ksi * 32, et * 16);
      c = wmma_f16(a, bd, c);
    }
    c_store(&ylds[0][0], INNERD, 0, h * DHH + et * 16, c, 1.0f);
  }
  __syncthreads();

  // out[16,128] tile: wave h produces output columns [h*16, h*16+16)
  v8f c = {};
#pragma unroll
  for (int ksi = 0; ksi < 16; ++ksi) {
    v16h a  = a_frag(&ylds[0][0], INNERD, 0, ksi * 32);
    v16h bw = bT_frag(Wo16, INNERD, ksi * 32, h * 16);
    c = wmma_f16(a, bw, c);
  }
  const int l = lane_id(); const int half = l >> 4; const int o = h * 16 + (l & 15);
  const float bias = bo[o];
  float* ob = out + ((size_t)b * NN + n0) * DD;
#pragma unroll
  for (int r = 0; r < 8; ++r)
    ob[(size_t)(half * 8 + r) * DD + o] = c[r] + bias;
}

extern "C" void kernel_launch(void* const* d_in, const int* in_sizes, int n_in,
                              void* d_out, int out_size, void* d_ws, size_t ws_size,
                              hipStream_t stream) {
  const float* queries = (const float*)d_in[0];
  const float* keys    = (const float*)d_in[1];
  const float* values  = (const float*)d_in[2];
  const float* Wq      = (const float*)d_in[3];
  const float* Wk      = (const float*)d_in[4];
  const float* Wo      = (const float*)d_in[5];
  const float* bo      = (const float*)d_in[6];
  float* out = (float*)d_out;

  // Workspace layout (1 MB total):
  //   [0,   512K) dotsT  f32 [B,H,64,64]
  //   [512K,768K) dots16 f16 [B,H,64,64] (1/N folded)
  //   [768K,896K) Wq16   f16 [512,128]
  //   [896K,  1M) Wo16   f16 [128,512]
  char* ws = (char*)d_ws;
  float*    dotsT  = (float*)ws;
  _Float16* dots16 = (_Float16*)(ws + 512 * 1024);
  _Float16* Wq16   = (_Float16*)(ws + 768 * 1024);
  _Float16* Wo16   = (_Float16*)(ws + 896 * 1024);

  const int ndots = BB * HH * DHH * DHH;      // 131072
  const int nw    = INNERD * DD;              // 65536

  hipLaunchKernelGGL(k_zero_f32, dim3((ndots + 255) / 256), dim3(256), 0, stream,
                     dotsT, ndots);
  hipLaunchKernelGGL(k_cvt_f16, dim3((nw / 4 + 255) / 256), dim3(256), 0, stream,
                     Wq, Wq16, nw / 4, 1.0f);
  hipLaunchKernelGGL(k_cvt_f16, dim3((nw / 4 + 255) / 256), dim3(256), 0, stream,
                     Wo, Wo16, nw / 4, 1.0f);
  hipLaunchKernelGGL(k_kv_dots, dim3(32, BB), dim3(256), 0, stream,
                     keys, values, Wk, dotsT);
  hipLaunchKernelGGL(k_cvt_f16, dim3((ndots / 4 + 255) / 256), dim3(256), 0, stream,
                     dotsT, dots16, ndots / 4, 1.0f / (float)NN);
  hipLaunchKernelGGL(k_out, dim3(NN / 16, BB), dim3(256), 0, stream,
                     queries, Wq16, dots16, Wo16, bo, out);
}
